// LstmRnn_66915590472224
// MI455X (gfx1250) — compile-verified, matrix operand order checked by reading, not run on hardware
//
#include <hip/hip_runtime.h>
#include <hip/hip_bf16.h>

// Problem constants (match reference)
#define BATCH 8192
#define TSEQ  48
#define FDIM  64
#define HDIM  128
#define OUTT  24
#define G4H   512   // 4*H

typedef __bf16 bf16_t;
typedef __attribute__((ext_vector_type(16))) __bf16 v16bf;
typedef __attribute__((ext_vector_type(8)))  __bf16 v8bf;
typedef __attribute__((ext_vector_type(8)))  float  v8f;

// Workspace offsets (in bf16 elements) for fragment-packed weights
#define OFF_W1  0
#define OFF_U1  (OFF_W1  + 64 * 512)
#define OFF_W2  (OFF_U1  + 128 * 512)
#define OFF_U2  (OFF_W2  + 64 * 512)
#define OFF_WD1 (OFF_U2  + 128 * 512)
#define OFF_WD  (OFF_WD1 + 128 * 128)

// ---------------------------------------------------------------------------
// Pre-pass: convert f32 weights [K,N] row-major -> bf16, swizzled into WMMA
// B-fragment layout: [nt][kt][lane][16 halves]; one lane's fragment is a
// contiguous 32B chunk. 16-bit B (32x16 KxN):
//   lane = nn + (kk&8 ? 16:0); half = (kk>>4)*8 + (kk&7)
// ---------------------------------------------------------------------------
__global__ void pack_weight_bf16(const float* __restrict__ src,
                                 bf16_t* __restrict__ dst, int K, int N) {
  int idx = blockIdx.x * 256 + threadIdx.x;
  if (idx >= K * N) return;
  int k = idx / N, n = idx - k * N;
  int kt = k >> 5, kk = k & 31;
  int nt = n >> 4, nn = n & 15;
  int lanep = nn + ((kk & 8) ? 16 : 0);
  int halfp = ((kk >> 4) << 3) + (kk & 7);
  int KT = K >> 5;
  dst[((size_t)(nt * KT + kt) * 32 + lanep) * 16 + halfp] =
      (bf16_t)src[(size_t)k * N + n];
}

// ---------------------------------------------------------------------------
// Fast gate nonlinearities on the hardware TRANS pipe (v_exp_f32 + v_rcp_f32).
// ~1 ulp, far below the bf16 operand quantization; co-executes with WMMA.
// ---------------------------------------------------------------------------
__device__ __forceinline__ float sigm(float x) {
  return __builtin_amdgcn_rcpf(1.0f + __expf(-x));
}
__device__ __forceinline__ float tanh_fast(float x) {
  return 2.0f * __builtin_amdgcn_rcpf(1.0f + __expf(-2.0f * x)) - 1.0f;
}

// Assemble a 16x32 bf16 A-fragment from row-major LDS (lda halves per row).
// Lanes 0-15: row=lane,    K = {k0..k0+7} u {k0+16..k0+23}
// Lanes 16-31: row=lane-16, K = {k0+8..k0+15} u {k0+24..k0+31}
__device__ __forceinline__ v16bf load_a(const bf16_t* sh, int lda, int k0, int lane) {
  int row  = lane & 15;
  int koff = k0 + ((lane & 16) >> 1);
  const bf16_t* p = sh + row * lda + koff;
  v8bf lo = *(const v8bf*)(p);
  v8bf hi = *(const v8bf*)(p + 16);
  v16bf a;
#pragma unroll
  for (int i = 0; i < 8; ++i) { a[i] = lo[i]; a[i + 8] = hi[i]; }
  return a;
}

__device__ __forceinline__ v8f wmma_bf16(v16bf a, v16bf b, v8f c) {
  return __builtin_amdgcn_wmma_f32_16x16x32_bf16(false, a, false, b, (short)0, c,
                                                 false, false);
}

// Generic wave GEMM: preload ALL A-fragments, then the full WMMA chain.
// B fragments are loop-invariant -> compiler pins them in VGPRs.
template <int NT, int NKT>
__device__ __forceinline__ void wgemm(v8f* acc, const bf16_t* shA, int lda,
                                      const v16bf* Bp, int ntbase, int lane) {
  v16bf a[NKT];
#pragma unroll
  for (int kt = 0; kt < NKT; ++kt) a[kt] = load_a(shA, lda, kt * 32, lane);
#pragma unroll
  for (int kt = 0; kt < NKT; ++kt)
#pragma unroll
    for (int j = 0; j < NT; ++j)
      acc[j] = wmma_bf16(a[kt], Bp[(size_t)((ntbase + j) * NKT + kt) * 32 + lane],
                         acc[j]);
}

// ---------------------------------------------------------------------------
// Fused LSTM step for one wave. Wave w owns hidden columns [32w, 32w+32):
// it computes the i/f/g/o tiles for those columns, so the gate nonlinearity
// runs entirely in registers (no z spill to LDS). Cell state cst[16] lives in
// registers across the whole 71-step recurrence.
// acc index = gate*2 + j ; n-tile = gate*8 + wave*2 + j
// ---------------------------------------------------------------------------
__device__ __forceinline__ void lstm_step_wave(
    const bf16_t* shX, const bf16_t* shHin, bf16_t* shHout,
    const v16bf* Wp /*KT=2*/, const v16bf* Up /*KT=4*/,
    const float* bI, const float* bF, const float* bG, const float* bO,
    float* cst, int wave, int lane) {
  // Preload all A fragments (x: K=64 -> 2, h: K=128 -> 4)
  v16bf ax[2], ah[4];
#pragma unroll
  for (int kt = 0; kt < 2; ++kt) ax[kt] = load_a(shX, FDIM, kt * 32, lane);
#pragma unroll
  for (int kt = 0; kt < 4; ++kt) ah[kt] = load_a(shHin, HDIM, kt * 32, lane);

  v8f acc[8] = {};
#pragma unroll
  for (int kt = 0; kt < 2; ++kt)
#pragma unroll
    for (int t8 = 0; t8 < 8; ++t8) {
      int nt = (t8 >> 1) * 8 + wave * 2 + (t8 & 1);
      acc[t8] = wmma_bf16(ax[kt], Wp[(size_t)(nt * 2 + kt) * 32 + lane], acc[t8]);
    }
#pragma unroll
  for (int kt = 0; kt < 4; ++kt)
#pragma unroll
    for (int t8 = 0; t8 < 8; ++t8) {
      int nt = (t8 >> 1) * 8 + wave * 2 + (t8 & 1);
      acc[t8] = wmma_bf16(ah[kt], Up[(size_t)(nt * 4 + kt) * 32 + lane], acc[t8]);
    }

  // In-register gate math; only the new h touches LDS.
  int rbase = (lane >> 4) << 3;
  int cl    = lane & 15;
#pragma unroll
  for (int j = 0; j < 2; ++j) {
    int col = wave * 32 + j * 16 + cl;
#pragma unroll
    for (int v = 0; v < 8; ++v) {
      float iv = sigm(acc[0 + j][v] + bI[j]);
      float fv = sigm(acc[2 + j][v] + bF[j]);
      float gv = tanh_fast(acc[4 + j][v] + bG[j]);
      float ov = sigm(acc[6 + j][v] + bO[j]);
      float cn = fv * cst[j * 8 + v] + iv * gv;
      cst[j * 8 + v] = cn;
      shHout[(rbase + v) * HDIM + col] = (bf16_t)(ov * tanh_fast(cn));
    }
  }
}

// relu(x @ Wd1 + bd1): each wave does 2 of 8 n-tiles (K=128)
__device__ __forceinline__ void dense_relu(const bf16_t* shIn, const v16bf* Wp,
                                           const float* __restrict__ bias,
                                           bf16_t* shOut, int wave, int lane) {
  v8f acc[2] = {};
  wgemm<2, 4>(acc, shIn, HDIM, Wp, wave * 2, lane);
  int rbase = (lane >> 4) << 3;
  int cl    = lane & 15;
#pragma unroll
  for (int j = 0; j < 2; ++j) {
    int c   = (wave * 2 + j) * 16 + cl;
    float b = bias[c];
#pragma unroll
    for (int v = 0; v < 8; ++v) {
      float x = acc[j][v] + b;
      x = x > 0.f ? x : 0.f;
      shOut[(rbase + v) * HDIM + c] = (bf16_t)x;
    }
  }
}

// pred = x @ Wd + bd -> global output + LDS feedback (decode input)
__device__ __forceinline__ void dense_out(const bf16_t* shIn, const v16bf* Wp,
                                          const float* __restrict__ bias,
                                          bf16_t* shX, float* __restrict__ out,
                                          int b0, int step, int wave, int lane) {
  v8f acc[1] = {};
  wgemm<1, 4>(acc, shIn, HDIM, Wp, wave, lane);
  int rbase = (lane >> 4) << 3;
  int cl    = lane & 15;
  int c     = wave * 16 + cl;
  float b   = bias[c];
#pragma unroll
  for (int v = 0; v < 8; ++v) {
    float x = acc[0][v] + b;
    int m   = rbase + v;
    out[((size_t)(b0 + m) * OUTT + step) * FDIM + c] = x;
    shX[m * FDIM + c] = (bf16_t)x;
  }
}

// ---------------------------------------------------------------------------
// Main persistent-recurrence kernel: one block = 16 batch rows, 4 waves.
// ---------------------------------------------------------------------------
__global__ void __launch_bounds__(128, 1)
lstm_seq2seq(const float* __restrict__ inputs,
             const float* __restrict__ b1, const float* __restrict__ b2,
             const float* __restrict__ bd1, const float* __restrict__ bd,
             const bf16_t* __restrict__ ws, float* __restrict__ out) {
  __shared__ bf16_t sh_h[2][16 * HDIM];  // h ping-pong (bf16 A-operand source)
  __shared__ bf16_t sh_d[2][16 * HDIM];  // head intermediates
  __shared__ bf16_t sh_x[16 * FDIM];     // current cell input (x_t or pred)

  const int tid  = threadIdx.x;
  const int lane = tid & 31;
  const int wave = tid >> 5;
  const int b0   = blockIdx.x * 16;

  const v16bf* W1p  = (const v16bf*)(ws + OFF_W1);
  const v16bf* U1p  = (const v16bf*)(ws + OFF_U1);
  const v16bf* W2p  = (const v16bf*)(ws + OFF_W2);
  const v16bf* U2p  = (const v16bf*)(ws + OFF_U2);
  const v16bf* Wd1p = (const v16bf*)(ws + OFF_WD1);
  const v16bf* Wdp  = (const v16bf*)(ws + OFF_WD);

  // Warm L2 with the recurrent weight regions (global_prefetch_b8)
  __builtin_prefetch(ws + OFF_U1 + (size_t)tid * 512, 0, 0);
  __builtin_prefetch(ws + OFF_U2 + (size_t)tid * 512, 0, 0);

  // Per-lane gate biases for this wave's hidden columns [32w, 32w+32)
  const int cl = lane & 15;
  float bI1[2], bF1[2], bG1[2], bO1[2];
  float bI2[2], bF2[2], bG2[2], bO2[2];
#pragma unroll
  for (int j = 0; j < 2; ++j) {
    int col = wave * 32 + j * 16 + cl;
    bI1[j] = b1[col];       bI2[j] = b2[col];
    bF1[j] = b1[128 + col]; bF2[j] = b2[128 + col];
    bG1[j] = b1[256 + col]; bG2[j] = b2[256 + col];
    bO1[j] = b1[384 + col]; bO2[j] = b2[384 + col];
  }

  // h0 = c0 = 0
  for (int i = tid; i < 16 * HDIM; i += 128) sh_h[0][i] = (bf16_t)0.0f;
  float cst[16];
#pragma unroll
  for (int r = 0; r < 16; ++r) cst[r] = 0.f;
  int cur = 0;

  // ---- warmup: 48 encoder steps, x_{t+1} prefetched into registers ----
  const int xr = tid >> 3, xc = (tid & 7) << 3;
  const float* xbase = inputs + ((size_t)(b0 + xr) * TSEQ) * FDIM + xc;
  float4 f0 = *(const float4*)(xbase);
  float4 f1 = *(const float4*)(xbase + 4);

#pragma unroll 1
  for (int t = 0; t < TSEQ; ++t) {
    {  // commit prefetched x_t -> LDS bf16, then issue loads for x_{t+1}
      bf16_t* dp = &sh_x[xr * FDIM + xc];
      dp[0] = (bf16_t)f0.x; dp[1] = (bf16_t)f0.y; dp[2] = (bf16_t)f0.z; dp[3] = (bf16_t)f0.w;
      dp[4] = (bf16_t)f1.x; dp[5] = (bf16_t)f1.y; dp[6] = (bf16_t)f1.z; dp[7] = (bf16_t)f1.w;
      if (t + 1 < TSEQ) {
        const float* xp = xbase + (size_t)(t + 1) * FDIM;
        f0 = *(const float4*)(xp);
        f1 = *(const float4*)(xp + 4);
      }
    }
    __syncthreads();
    lstm_step_wave(sh_x, sh_h[cur], sh_h[cur ^ 1], W1p, U1p,
                   bI1, bF1, bG1, bO1, cst, wave, lane);
    cur ^= 1;
    __syncthreads();
  }

  // ---- head for pred0 + autoregressive decode ----
#pragma unroll 1
  for (int s = 0; s < OUTT; ++s) {
    if (s > 0) {  // decode LSTM step with x = previous pred
      lstm_step_wave(sh_x, sh_h[cur], sh_h[cur ^ 1], W2p, U2p,
                     bI2, bF2, bG2, bO2, cst, wave, lane);
      cur ^= 1;
      __syncthreads();
    }
    // head: relu(relu(h@Wd1+bd1)@Wd1+bd1)@Wd + bd
    dense_relu(sh_h[cur], Wd1p, bd1, sh_d[0], wave, lane);
    __syncthreads();
    dense_relu(sh_d[0], Wd1p, bd1, sh_d[1], wave, lane);
    __syncthreads();
    dense_out(sh_d[1], Wdp, bd, sh_x, out, b0, s, wave, lane);
    __syncthreads();
  }
}

// ---------------------------------------------------------------------------
extern "C" void kernel_launch(void* const* d_in, const int* in_sizes, int n_in,
                              void* d_out, int out_size, void* d_ws, size_t ws_size,
                              hipStream_t stream) {
  (void)in_sizes; (void)n_in; (void)out_size; (void)ws_size;
  const float* inputs = (const float*)d_in[0];
  const float* W1  = (const float*)d_in[1];
  const float* U1  = (const float*)d_in[2];
  const float* b1  = (const float*)d_in[3];
  const float* W2  = (const float*)d_in[4];
  const float* U2  = (const float*)d_in[5];
  const float* b2  = (const float*)d_in[6];
  const float* Wd1 = (const float*)d_in[7];
  const float* bd1 = (const float*)d_in[8];
  const float* Wd  = (const float*)d_in[9];
  const float* bd  = (const float*)d_in[10];
  bf16_t* ws = (bf16_t*)d_ws;

  auto pack = [&](const float* s, size_t off, int K, int N) {
    int n = K * N;
    pack_weight_bf16<<<(n + 255) / 256, 256, 0, stream>>>(s, ws + off, K, N);
  };
  pack(W1, OFF_W1, 64, 512);
  pack(U1, OFF_U1, 128, 512);
  pack(W2, OFF_W2, 64, 512);
  pack(U2, OFF_U2, 128, 512);
  pack(Wd1, OFF_WD1, 128, 128);
  pack(Wd, OFF_WD, 128, 64);

  lstm_seq2seq<<<BATCH / 16, 128, 0, stream>>>(inputs, b1, b2, bd1, bd, ws,
                                               (float*)d_out);
}
